// RotatedFasterRCNNRoIHead_82566451298942
// MI455X (gfx1250) — matrix operand order repaired and checked by reading, not compiled, not run on hardware
//
#include <hip/hip_runtime.h>
#include <hip/hip_bf16.h>

typedef __attribute__((ext_vector_type(16))) _Float16 v16h;
typedef __attribute__((ext_vector_type(8)))  float    v8f;

#define NEGV     (-1.0e30f)
#define CLAMPV   (4.135166556742356f)   /* log(1000/16) */
#define NPROP    1000
#define NIMG     2
#define MROWS    2000                   /* B*N */
#define K1DIM    12544                  /* 256*49 */
#define HDIM     1024
#define NCAND    15000                  /* N*(NC-1) */
#define TOPKN    512
#define NDETS    100

#define ASTRIDE  40                     /* padded LDS row strides (f16 units) */
#define BSTRIDE  40

union FragH { v16h h; uint4 q[2]; };

// ---------------------------------------------------------------------------
// f32 -> f16 weight conversion
// ---------------------------------------------------------------------------
__global__ void f32_to_f16_kernel(const float* __restrict__ src,
                                  _Float16* __restrict__ dst, int n) {
  int i = blockIdx.x * blockDim.x + threadIdx.x;
  if (i < n) dst[i] = (_Float16)src[i];
}

// ---------------------------------------------------------------------------
// Rotated RoI-align: one block per (b,n); pooled[m][c*49+ph*7+pw] in f16
// ---------------------------------------------------------------------------
__global__ void roi_align_kernel(const float* __restrict__ feat,
                                 const float* __restrict__ props,
                                 _Float16* __restrict__ pooled) {
  int bn = blockIdx.x;                   // 0..1999
  int b  = bn / NPROP;
  const float* p = props + (size_t)bn * 5;
  __shared__ int   sx0[49], sy0[49];
  __shared__ float sfx[49], sfy[49];
  int tid = threadIdx.x;
  if (tid < 49) {
    int ph = tid / 7, pw = tid % 7;
    float cx = p[0], cy = p[1], w = p[2], h = p[3], a = p[4];
    float ct = cosf(a), st = sinf(a);
    float fw = (pw + 0.5f) / 7.0f - 0.5f;
    float fh = (ph + 0.5f) / 7.0f - 0.5f;
    float lx = w * fw, ly = h * fh;
    float px = (cx + lx * ct - ly * st) * 0.125f;
    float py = (cy + lx * st + ly * ct) * 0.125f;
    px = fminf(fmaxf(px, 0.0f), 99.0f);
    py = fminf(fmaxf(py, 0.0f), 99.0f);
    int x0 = (int)floorf(px); x0 = min(max(x0, 0), 98);
    int y0 = (int)floorf(py); y0 = min(max(y0, 0), 98);
    sx0[tid] = x0; sy0[tid] = y0;
    sfx[tid] = px - (float)x0; sfy[tid] = py - (float)y0;
  }
  __syncthreads();
  const float* fimg = feat + (size_t)b * 256 * 10000;
  _Float16* out = pooled + (size_t)bn * K1DIM;
  for (int idx = tid; idx < K1DIM; idx += blockDim.x) {
    int c = idx / 49, pp = idx % 49;
    int x0 = sx0[pp], y0 = sy0[pp];
    float fx = sfx[pp], fy = sfy[pp];
    const float* fc = fimg + (size_t)c * 10000;
    float v00 = fc[y0 * 100 + x0];
    float v01 = fc[y0 * 100 + x0 + 1];
    float v10 = fc[(y0 + 1) * 100 + x0];
    float v11 = fc[(y0 + 1) * 100 + x0 + 1];
    float v = v00 * (1 - fy) * (1 - fx) + v01 * (1 - fy) * fx +
              v10 * fy * (1 - fx) + v11 * fy * fx;
    out[idx] = (_Float16)v;
  }
}

// ---------------------------------------------------------------------------
// WMMA GEMM: C = ReLU(A[f16 MxK] * W[f16 KxN] + bias), out f16 or f32.
// 128 threads = 4 waves; block tile 64x64; wave tile 32x32 (2x2 WMMA 16x16).
// A tile staged with GLOBAL_LOAD_ASYNC_TO_LDS_B128 (ASYNCcnt path);
// B tile stored TRANSPOSED in LDS so every fragment is 2x ds_load_b128.
// ---------------------------------------------------------------------------
__global__ void wmma_gemm_relu_kernel(const _Float16* __restrict__ A,
                                      const _Float16* __restrict__ W,
                                      const float* __restrict__ bias,
                                      float* __restrict__ Cf,
                                      _Float16* __restrict__ Ch,
                                      int M, int K, int N, int out_f16) {
  __shared__ _Float16 As[64 * ASTRIDE];   // As[row][k], k = 0..31
  __shared__ _Float16 Bt[64 * BSTRIDE];   // Bt[n][k],  k = 0..31 (transposed)
  const int tid  = threadIdx.x;
  const int wave = tid >> 5;
  const int lane = tid & 31;
  const int bm = blockIdx.x * 64;
  const int bn = blockIdx.y * 64;
  const int wm = (wave >> 1) * 32;
  const int wn = (wave & 1) * 32;
  const int lr = lane & 15;
  const int g  = lane >> 4;

  v8f c00 = {}, c01 = {}, c10 = {}, c11 = {};

  // precomputed staging coordinates
  const int arow0 = tid >> 2,        acol0 = (tid & 3) * 8;
  const int arow1 = (tid + 128) >> 2, acol1 = ((tid + 128) & 3) * 8;
  const int brow0 = tid >> 3,        bcol0 = (tid & 7) * 8;
  const int brow1 = (tid + 128) >> 3, bcol1 = ((tid + 128) & 7) * 8;
  int agr0 = bm + arow0; if (agr0 >= M) agr0 = M - 1;
  int agr1 = bm + arow1; if (agr1 >= M) agr1 = M - 1;

  // loop-invariant LDS destinations for the async A copies
  // (LDS aperture: low 32 bits of the generic address are the LDS byte offset)
  const unsigned lA0 = (unsigned)(size_t)(const void*)&As[arow0 * ASTRIDE + acol0];
  const unsigned lA1 = (unsigned)(size_t)(const void*)&As[arow1 * ASTRIDE + acol1];

  for (int k0 = 0; k0 < K; k0 += 32) {
    // ---- stage A tile 64x32 f16 via async global->LDS B128 copies ----
    {
      const _Float16* g0 = A + (size_t)agr0 * K + k0 + acol0;
      const _Float16* g1 = A + (size_t)agr1 * K + k0 + acol1;
      asm volatile("global_load_async_to_lds_b128 %0, %1, off"
                   :: "v"(lA0), "v"(g0) : "memory");
      asm volatile("global_load_async_to_lds_b128 %0, %1, off"
                   :: "v"(lA1), "v"(g1) : "memory");
    }
    // ---- stage B tile 32x64 f16, transposed into Bt[n][k] ----
    {
      uint4 v0 = *(const uint4*)(W + (size_t)(k0 + brow0) * N + bn + bcol0);
      uint4 v1 = *(const uint4*)(W + (size_t)(k0 + brow1) * N + bn + bcol1);
      const _Float16* h0 = (const _Float16*)&v0;
      const _Float16* h1 = (const _Float16*)&v1;
#pragma unroll
      for (int j = 0; j < 8; ++j) Bt[(bcol0 + j) * BSTRIDE + brow0] = h0[j];
#pragma unroll
      for (int j = 0; j < 8; ++j) Bt[(bcol1 + j) * BSTRIDE + brow1] = h1[j];
    }
    // prefetch next K-step tiles while this one is consumed
    if (k0 + 32 < K) {
      __builtin_prefetch(A + (size_t)agr0 * K + (k0 + 32) + acol0, 0, 3);
      __builtin_prefetch(W + (size_t)(k0 + 32 + brow0) * N + bn + bcol0, 0, 3);
    }
    // all of this wave's async LDS writes must land before we pass the barrier
    asm volatile("s_wait_asynccnt 0x0" ::: "memory");
    __syncthreads();

    // ---- assemble fragments: every one is two contiguous 16B LDS loads ----
    FragH a0, a1, b0, b1;
    {
      const _Float16* r0 = &As[(wm + lr) * ASTRIDE + 8 * g];
      a0.q[0] = *(const uint4*)(r0);
      a0.q[1] = *(const uint4*)(r0 + 16);
      const _Float16* r1 = &As[(wm + 16 + lr) * ASTRIDE + 8 * g];
      a1.q[0] = *(const uint4*)(r1);
      a1.q[1] = *(const uint4*)(r1 + 16);
      const _Float16* c0 = &Bt[(wn + lr) * BSTRIDE + 8 * g];
      b0.q[0] = *(const uint4*)(c0);
      b0.q[1] = *(const uint4*)(c0 + 16);
      const _Float16* c1 = &Bt[(wn + 16 + lr) * BSTRIDE + 8 * g];
      b1.q[0] = *(const uint4*)(c1);
      b1.q[1] = *(const uint4*)(c1 + 16);
    }
    c00 = __builtin_amdgcn_wmma_f32_16x16x32_f16(false, a0.h, false, b0.h, (short)0, c00, false, false);
    c01 = __builtin_amdgcn_wmma_f32_16x16x32_f16(false, a0.h, false, b1.h, (short)0, c01, false, false);
    c10 = __builtin_amdgcn_wmma_f32_16x16x32_f16(false, a1.h, false, b0.h, (short)0, c10, false, false);
    c11 = __builtin_amdgcn_wmma_f32_16x16x32_f16(false, a1.h, false, b1.h, (short)0, c11, false, false);
    __syncthreads();
  }

  // ---- epilogue: C layout VGPR i -> (M = i + 8*g, N = lane&15) ----
  v8f accs[4] = {c00, c01, c10, c11};
  const int rb[4] = {bm + wm, bm + wm, bm + wm + 16, bm + wm + 16};
  const int cb[4] = {bn + wn, bn + wn + 16, bn + wn, bn + wn + 16};
  const bool full = (bm + 64 <= M);
#pragma unroll
  for (int t = 0; t < 4; ++t) {
    int col = cb[t] + lr;
    float bv = bias[col];
    int rbase = rb[t] + 8 * g;
    if (full) {
#pragma unroll
      for (int i = 0; i < 8; ++i) {
        float v = accs[t][i] + bv;
        v = v > 0.0f ? v : 0.0f;
        size_t o = (size_t)(rbase + i) * N + col;
        if (out_f16) Ch[o] = (_Float16)v;
        else         Cf[o] = v;
      }
    } else {
#pragma unroll
      for (int i = 0; i < 8; ++i) {
        int row = rbase + i;
        if (row < M) {
          float v = accs[t][i] + bv;
          v = v > 0.0f ? v : 0.0f;
          size_t o = (size_t)row * N + col;
          if (out_f16) Ch[o] = (_Float16)v;
          else         Cf[o] = v;
        }
      }
    }
  }
}

// ---------------------------------------------------------------------------
// Heads: logits(16)+reg(80), softmax, decode, threshold -> fb / sc
// ---------------------------------------------------------------------------
__global__ void heads_kernel(const float* __restrict__ x2,
                             const float* __restrict__ cls_w,
                             const float* __restrict__ cls_b,
                             const float* __restrict__ reg_w,
                             const float* __restrict__ reg_b,
                             const float* __restrict__ props,
                             float* __restrict__ fbx,
                             float* __restrict__ sc) {
  int m = blockIdx.x;                 // 0..1999
  int b = m / NPROP, n = m % NPROP;
  __shared__ float xs[HDIM];
  __shared__ float lg[16];
  __shared__ float rg[80];
  int tid = threadIdx.x;              // 128
  for (int i = tid; i < HDIM; i += 128) xs[i] = x2[(size_t)m * HDIM + i];
  __syncthreads();
  if (tid < 16) {
    float s = cls_b[tid];
    for (int k = 0; k < HDIM; ++k) s += xs[k] * cls_w[k * 16 + tid];
    lg[tid] = s;
  } else if (tid < 96) {
    int j = tid - 16;
    float s = reg_b[j];
    for (int k = 0; k < HDIM; ++k) s += xs[k] * reg_w[k * 80 + j];
    rg[j] = s;
  }
  __syncthreads();
  if (tid < 15) {
    int c = tid + 1;
    float mx = lg[0];
    for (int i = 1; i < 16; ++i) mx = fmaxf(mx, lg[i]);
    float sum = 0.0f;
    for (int i = 0; i < 16; ++i) sum += expf(lg[i] - mx);
    float prob = expf(lg[c] - mx) / sum;
    const float* p = props + (size_t)m * 5;
    float dx = rg[c * 5 + 0] * 0.1f;
    float dy = rg[c * 5 + 1] * 0.1f;
    float dw = fminf(rg[c * 5 + 2] * 0.2f, CLAMPV);
    float dh = fminf(rg[c * 5 + 3] * 0.2f, CLAMPV);
    float da = rg[c * 5 + 4] * 0.1f;
    float cx = dx * p[2] + p[0];
    float cy = dy * p[3] + p[1];
    float w  = p[2] * expf(dw);
    float h  = p[3] * expf(dh);
    float an = p[4] + da;
    int j = n * 15 + (c - 1);
    float* dst = fbx + ((size_t)b * NCAND + j) * 5;
    dst[0] = cx; dst[1] = cy; dst[2] = w; dst[3] = h; dst[4] = an;
    bool valid = (prob > 0.05f) && (w >= 0.01f) && (h >= 0.01f);
    sc[(size_t)b * NCAND + j] = valid ? prob : NEGV;
  }
}

// ---------------------------------------------------------------------------
// Top-512 per image via 512 block-wide argmax sweeps (descending, stable)
// ---------------------------------------------------------------------------
__global__ void topk_kernel(const float* __restrict__ sc,
                            const float* __restrict__ fbx,
                            float* __restrict__ scm,
                            float* __restrict__ tb,
                            float* __restrict__ ts,
                            int*   __restrict__ tl) {
  int b = blockIdx.x;
  int tid = threadIdx.x;              // 1024
  const float* s = sc + (size_t)b * NCAND;
  float* sm = scm + (size_t)b * NCAND;
  for (int i = tid; i < NCAND; i += 1024) sm[i] = s[i];
  __syncthreads();
  __shared__ float rv[1024];
  __shared__ int   ri[1024];
  for (int k = 0; k < TOPKN; ++k) {
    float bv = -3.0e38f; int bi = 0x7fffffff;
    for (int i = tid; i < NCAND; i += 1024) {
      float v = sm[i];
      if (v > bv || (v == bv && i < bi)) { bv = v; bi = i; }
    }
    rv[tid] = bv; ri[tid] = bi;
    __syncthreads();
    for (int st = 512; st > 0; st >>= 1) {
      if (tid < st) {
        float vo = rv[tid + st]; int io = ri[tid + st];
        if (vo > rv[tid] || (vo == rv[tid] && io < ri[tid])) { rv[tid] = vo; ri[tid] = io; }
      }
      __syncthreads();
    }
    if (tid == 0) {
      int i = ri[0];
      ts[(size_t)b * TOPKN + k] = rv[0];
      tl[(size_t)b * TOPKN + k] = 1 + (i % 15);
      const float* src = fbx + ((size_t)b * NCAND + i) * 5;
      float* dst = tb + ((size_t)b * TOPKN + k) * 5;
      for (int q = 0; q < 5; ++q) dst[q] = src[q];
      sm[i] = NEGV;
    }
    __syncthreads();
  }
}

// ---------------------------------------------------------------------------
// Rotated IoU matrix (reference-faithful masked hull formulation)
// ---------------------------------------------------------------------------
__device__ __forceinline__ void corners_of(const float* bx, float* X, float* Y) {
  float cx = bx[0], cy = bx[1], w = bx[2], h = bx[3], a = bx[4];
  float c = cosf(a), s = sinf(a);
  const float lx[4] = {-0.5f * w, 0.5f * w, 0.5f * w, -0.5f * w};
  const float ly[4] = {-0.5f * h, -0.5f * h, 0.5f * h, 0.5f * h};
#pragma unroll
  for (int q = 0; q < 4; ++q) {
    X[q] = cx + lx[q] * c - ly[q] * s;
    Y[q] = cy + lx[q] * s + ly[q] * c;
  }
}

__device__ __forceinline__ bool pt_in(float px, float py, const float* bx) {
  float c = cosf(bx[4]), s = sinf(bx[4]);
  float rx = px - bx[0], ry = py - bx[1];
  float lx = rx * c + ry * s;
  float ly = -rx * s + ry * c;
  return (fabsf(lx) <= 0.5f * bx[2] + 1e-5f) && (fabsf(ly) <= 0.5f * bx[3] + 1e-5f);
}

__global__ void iou_kernel(const float* __restrict__ tb,
                           const int* __restrict__ tl,
                           float* __restrict__ iou) {
  int gid = blockIdx.x * blockDim.x + threadIdx.x;
  if (gid >= NIMG * TOPKN * TOPKN) return;
  int b = gid / (TOPKN * TOPKN);
  int r = gid % (TOPKN * TOPKN);
  int i = r / TOPKN, j = r % TOPKN;
  float out = 0.0f;
  if (tl[b * TOPKN + i] == tl[b * TOPKN + j]) {
    const float* bi = tb + ((size_t)b * TOPKN + i) * 5;
    const float* bj = tb + ((size_t)b * TOPKN + j) * 5;
    float Xi[4], Yi[4], Xj[4], Yj[4];
    corners_of(bi, Xi, Yi);
    corners_of(bj, Xj, Yj);
    float px[24], py[24];
    bool  m[24];
    int t2 = 0;
    for (int e1 = 0; e1 < 4; ++e1) {
      for (int e2 = 0; e2 < 4; ++e2) {
        float p1x = Xi[e1], p1y = Yi[e1];
        float d1x = Xi[(e1 + 1) & 3] - p1x, d1y = Yi[(e1 + 1) & 3] - p1y;
        float q1x = Xj[e2], q1y = Yj[e2];
        float d2x = Xj[(e2 + 1) & 3] - q1x, d2y = Yj[(e2 + 1) & 3] - q1y;
        float den = d1x * d2y - d1y * d2x;
        float ds  = (fabsf(den) < 1e-9f) ? 1.0f : den;
        float wx = q1x - p1x, wy = q1y - p1y;
        float t = (wx * d2y - wy * d2x) / ds;
        float u = (wx * d1y - wy * d1x) / ds;
        bool ok = (fabsf(den) > 1e-9f) && (t >= 0.0f) && (t <= 1.0f) &&
                  (u >= 0.0f) && (u <= 1.0f);
        px[t2] = p1x + t * d1x;
        py[t2] = p1y + t * d1y;
        m[t2] = ok;
        ++t2;
      }
    }
#pragma unroll
    for (int q = 0; q < 4; ++q) { px[16 + q] = Xi[q]; py[16 + q] = Yi[q]; m[16 + q] = pt_in(Xi[q], Yi[q], bj); }
#pragma unroll
    for (int q = 0; q < 4; ++q) { px[20 + q] = Xj[q]; py[20 + q] = Yj[q]; m[20 + q] = pt_in(Xj[q], Yj[q], bi); }
    int cnt = 0; float cxs = 0.0f, cys = 0.0f;
    for (int q = 0; q < 24; ++q) if (m[q]) { ++cnt; cxs += px[q]; cys += py[q]; }
    float cden = (float)(cnt >= 1 ? cnt : 1);
    float cenx = cxs / cden, ceny = cys / cden;
    float rx[24], ry[24], ang[24];
    for (int q = 0; q < 24; ++q) {
      rx[q]  = m[q] ? (px[q] - cenx) : 0.0f;
      ry[q]  = m[q] ? (py[q] - ceny) : 0.0f;
      ang[q] = atan2f(ry[q], rx[q]);
    }
    // stable insertion sort by angle (matches jnp.argsort stability)
    for (int a2 = 1; a2 < 24; ++a2) {
      float ka = ang[a2], xa = rx[a2], ya = ry[a2];
      int bI = a2 - 1;
      while (bI >= 0 && ang[bI] > ka) {
        ang[bI + 1] = ang[bI]; rx[bI + 1] = rx[bI]; ry[bI + 1] = ry[bI];
        --bI;
      }
      ang[bI + 1] = ka; rx[bI + 1] = xa; ry[bI + 1] = ya;
    }
    float ssum = 0.0f;
    for (int q = 0; q < 24; ++q) {
      int qn = (q + 1) % 24;
      ssum += rx[q] * ry[qn] - ry[q] * rx[qn];
    }
    float inter = 0.5f * fabsf(ssum);
    float ai = bi[2] * bi[3], aj = bj[2] * bj[3];
    out = inter / (ai + aj - inter + 1e-7f);
  }
  iou[gid] = out;
}

// ---------------------------------------------------------------------------
// NMS suppression scan (100 iters) + output gather
// ---------------------------------------------------------------------------
__global__ void nms_out_kernel(const float* __restrict__ tb,
                               const float* __restrict__ ts,
                               const int*   __restrict__ tl,
                               const float* __restrict__ iou,
                               float* __restrict__ out) {
  int b = blockIdx.x;
  int tid = threadIdx.x;               // 128
  __shared__ float scb[TOPKN];
  __shared__ float rv[128];
  __shared__ int   ri[128];
  __shared__ int   sidx[NDETS];
  __shared__ int   sval[NDETS];
  for (int i = tid; i < TOPKN; i += 128) scb[i] = ts[b * TOPKN + i];
  __syncthreads();
  for (int k = 0; k < NDETS; ++k) {
    float bv = -3.0e38f; int bi = 0x7fffffff;
    for (int i = tid; i < TOPKN; i += 128) {
      float v = scb[i];
      if (v > bv || (v == bv && i < bi)) { bv = v; bi = i; }
    }
    rv[tid] = bv; ri[tid] = bi;
    __syncthreads();
    for (int st = 64; st > 0; st >>= 1) {
      if (tid < st) {
        float vo = rv[tid + st]; int io = ri[tid + st];
        if (vo > rv[tid] || (vo == rv[tid] && io < ri[tid])) { rv[tid] = vo; ri[tid] = io; }
      }
      __syncthreads();
    }
    int i = ri[0];
    if (tid == 0) {
      sidx[k] = i;
      sval[k] = (rv[0] > 0.5f * NEGV) ? 1 : 0;
    }
    const float* row = iou + ((size_t)b * TOPKN + i) * TOPKN;
    for (int j = tid; j < TOPKN; j += 128)
      if (row[j] > 0.5f) scb[j] = NEGV;
    __syncthreads();
    if (tid == 0) scb[i] = NEGV;
    __syncthreads();
  }
  // write: ob (B,100,5) @0, os (B,100) @1000, ol (B,100) @1200
  for (int k = tid; k < NDETS; k += 128) {
    int idx = sidx[k], keep = sval[k];
    const float* bx = tb + ((size_t)b * TOPKN + idx) * 5;
    float* ob = out + (size_t)b * NDETS * 5 + k * 5;
    for (int q = 0; q < 5; ++q) ob[q] = keep ? bx[q] : 0.0f;
    out[NIMG * NDETS * 5 + b * NDETS + k] = keep ? ts[b * TOPKN + idx] : -1.0f;
    out[NIMG * NDETS * 5 + NIMG * NDETS + b * NDETS + k] =
        keep ? (float)tl[b * TOPKN + idx] : -1.0f;
  }
}

// ---------------------------------------------------------------------------
extern "C" void kernel_launch(void* const* d_in, const int* in_sizes, int n_in,
                              void* d_out, int out_size, void* d_ws, size_t ws_size,
                              hipStream_t stream) {
  (void)in_sizes; (void)n_in; (void)out_size; (void)ws_size;
  const float* features  = (const float*)d_in[0];
  const float* proposals = (const float*)d_in[1];
  const float* fc1_w = (const float*)d_in[2];
  const float* fc1_b = (const float*)d_in[3];
  const float* fc2_w = (const float*)d_in[4];
  const float* fc2_b = (const float*)d_in[5];
  const float* cls_w = (const float*)d_in[6];
  const float* cls_b = (const float*)d_in[7];
  const float* reg_w = (const float*)d_in[8];
  const float* reg_b = (const float*)d_in[9];
  float* out = (float*)d_out;

  char* ws = (char*)d_ws;
  size_t off = 0;
  auto take = [&](size_t bytes) -> char* {
    char* p = ws + off;
    off += (bytes + 255) & ~(size_t)255;
    return p;
  };
  _Float16* pooled = (_Float16*)take((size_t)MROWS * K1DIM * 2);
  _Float16* w1h    = (_Float16*)take((size_t)K1DIM * HDIM * 2);
  _Float16* w2h    = (_Float16*)take((size_t)HDIM * HDIM * 2);
  _Float16* x1h    = (_Float16*)take((size_t)MROWS * HDIM * 2);
  float*    x2f    = (float*)take((size_t)MROWS * HDIM * 4);
  float*    fbx    = (float*)take((size_t)NIMG * NCAND * 5 * 4);
  float*    sc     = (float*)take((size_t)NIMG * NCAND * 4);
  float*    scm    = (float*)take((size_t)NIMG * NCAND * 4);
  float*    tb     = (float*)take((size_t)NIMG * TOPKN * 5 * 4);
  float*    tsb    = (float*)take((size_t)NIMG * TOPKN * 4);
  int*      tlb    = (int*)take((size_t)NIMG * TOPKN * 4);
  float*    ioub   = (float*)take((size_t)NIMG * TOPKN * TOPKN * 4);

  int n1 = K1DIM * HDIM;
  f32_to_f16_kernel<<<(n1 + 255) / 256, 256, 0, stream>>>(fc1_w, w1h, n1);
  int n2 = HDIM * HDIM;
  f32_to_f16_kernel<<<(n2 + 255) / 256, 256, 0, stream>>>(fc2_w, w2h, n2);

  roi_align_kernel<<<MROWS, 256, 0, stream>>>(features, proposals, pooled);

  dim3 gemm_grid((MROWS + 63) / 64, HDIM / 64);
  wmma_gemm_relu_kernel<<<gemm_grid, 128, 0, stream>>>(
      pooled, w1h, fc1_b, nullptr, x1h, MROWS, K1DIM, HDIM, 1);
  wmma_gemm_relu_kernel<<<gemm_grid, 128, 0, stream>>>(
      x1h, w2h, fc2_b, x2f, nullptr, MROWS, HDIM, HDIM, 0);

  heads_kernel<<<MROWS, 128, 0, stream>>>(x2f, cls_w, cls_b, reg_w, reg_b,
                                          proposals, fbx, sc);

  topk_kernel<<<NIMG, 1024, 0, stream>>>(sc, fbx, scm, tb, tsb, tlb);

  int npairs = NIMG * TOPKN * TOPKN;
  iou_kernel<<<(npairs + 255) / 256, 256, 0, stream>>>(tb, tlb, ioub);

  nms_out_kernel<<<NIMG, 128, 0, stream>>>(tb, tsb, tlb, ioub, out);
}
